// ScannedRNN_15479062134960
// MI455X (gfx1250) — matrix operand order repaired and checked by reading, not compiled, not run on hardware
//
#include <hip/hip_runtime.h>
#include <hip/hip_bf16.h>
#include <math.h>

typedef __bf16 bf16;
typedef __attribute__((ext_vector_type(16))) __bf16 v16bf;
typedef __attribute__((ext_vector_type(8)))  float   v8f;

#define T_STEPS 512
#define B_DIM   128
#define H_DIM   512
#define NG      1536            // 3H
#define KTOT    1024            // H (x part) + H (h part)
#define NWG     16
#define WG_THREADS 256
#define JCOLS   32              // h columns owned per WG
#define LDS_ROWSTRIDE 1032      // 1024 + 8 bf16 pad -> conflict-free ds_load_b128
#define LDS_BYTES (96 * LDS_ROWSTRIDE * 2)
#define YS_OFF  ((size_t)B_DIM * H_DIM)   // d_out: [h_final (B,H)] [ys (T,B,H)]

struct alignas(16) bfx8 { bf16 v[8]; };
struct alignas(16) f32x4 { float v[4]; };

__device__ __forceinline__ float sigf(float x) { return 1.0f / (1.0f + __expf(-x)); }

// ---------------------------------------------------------------------------
// One-time pack: Wt[n][k] = bf16( k<512 ? Wi[k][n] : Wh[k-512][n] ), n in [0,1536)
// Also zero both h ping-pong buffers and the barrier counter (every call).
// ---------------------------------------------------------------------------
__global__ void gru_pack_kernel(const float* __restrict__ Wi,
                                const float* __restrict__ Wh,
                                bf16* __restrict__ Wt,
                                bf16* __restrict__ h0, bf16* __restrict__ h1,
                                unsigned* __restrict__ ctr) {
    size_t gid = (size_t)blockIdx.x * blockDim.x + threadIdx.x;
    size_t stride = (size_t)gridDim.x * blockDim.x;
    for (size_t i = gid; i < (size_t)NG * KTOT; i += stride) {
        int n = (int)(i / KTOT);
        int k = (int)(i % KTOT);
        float v = (k < H_DIM) ? Wi[(size_t)k * NG + n]
                              : Wh[(size_t)(k - H_DIM) * NG + n];
        Wt[i] = (bf16)v;
    }
    for (size_t i = gid; i < (size_t)B_DIM * H_DIM; i += stride) {
        h0[i] = (bf16)0.0f;
        h1[i] = (bf16)0.0f;
    }
    if (gid == 0) *ctr = 0u;
}

// ---------------------------------------------------------------------------
// Persistent GRU scan. 16 WGs x 256 threads (8 wave32 each).
//   WG g owns h columns [32g, 32g+32); wave w owns batch rows [16w, 16w+16).
//   Per step, per wave: D(16x96) += A(16x1024) * B(1024x96) via 192 bf16 WMMAs.
//   B slice lives in LDS (loaded once); h state ping-pongs through L2 in bf16.
// ---------------------------------------------------------------------------
__global__ void __launch_bounds__(WG_THREADS)
gru_scan_kernel(const float* __restrict__ x,
                const unsigned char* __restrict__ resets,
                const float* __restrict__ bi,
                const float* __restrict__ bh_n,
                const bf16* __restrict__ Wt,
                bf16* __restrict__ h0, bf16* __restrict__ h1,
                unsigned* __restrict__ ctr,
                float* __restrict__ out) {
    extern __shared__ bf16 ldsB[];

    const int tid  = threadIdx.x;
    const int g    = blockIdx.x;
    const int w    = tid >> 5;
    const int lane = tid & 31;
    const int l16  = lane & 15;
    const int lh   = lane >> 4;          // 0 or 1
    const int jbase = g * JCOLS;

    // ---- Stage this WG's 96x1024 bf16 weight slice into LDS (once) --------
    // LDS rows: [0,32)=r-gate cols jbase..+31, [32,64)=z, [64,96)=n.
    for (int c = tid; c < 96 * (KTOT / 8); c += WG_THREADS) {
        int lr = c / (KTOT / 8);
        int kc = (c % (KTOT / 8)) * 8;
        int gate = lr / JCOLS, sub = lr % JCOLS;
        size_t n = (size_t)gate * H_DIM + jbase + sub;
        *(bfx8*)&ldsB[(size_t)lr * LDS_ROWSTRIDE + kc] =
            *(const bfx8*)&Wt[n * KTOT + kc];
    }
    __syncthreads();

    // ---- Per-lane constants ----------------------------------------------
    const int j0 = jbase + l16;                       // j for s=0 sub-tile
    const float bi_r[2] = { bi[j0],             bi[j0 + 16] };
    const float bi_z[2] = { bi[H_DIM + j0],     bi[H_DIM + j0 + 16] };
    const float bi_n[2] = { bi[2*H_DIM + j0],   bi[2*H_DIM + j0 + 16] };
    const float bhn[2]  = { bh_n[j0],           bh_n[j0 + 16] };

    const int mA = 16 * w + l16;                      // A-operand batch row
    float hc[2][8];                                   // register carry h
    #pragma unroll
    for (int s = 0; s < 2; ++s)
        #pragma unroll
        for (int v = 0; v < 8; ++v) hc[s][v] = 0.0f;

    const bf16 zb = (bf16)0.0f;

    for (int t = 0; t < T_STEPS; ++t) {
        const bf16* __restrict__ hrd = (t & 1) ? h1 : h0;   // state entering t
        bf16* __restrict__       hwr = (t & 1) ? h0 : h1;   // state leaving t

        const unsigned char* rst = resets + (size_t)t * B_DIM;
        const bool rA = rst[mA] != 0;                 // mask for A-operand row
        unsigned long long mbits;                     // masks for D rows (v=0..7)
        __builtin_memcpy(&mbits, rst + 16 * w + 8 * lh, 8);

        if (t + 1 < T_STEPS) {                        // warm next x_t rows
            const float* nx = x + ((size_t)(t + 1) * B_DIM + mA) * H_DIM;
            __builtin_prefetch(nx,        0, 3);
            __builtin_prefetch(nx + 128,  0, 3);
            __builtin_prefetch(nx + 256,  0, 3);
            __builtin_prefetch(nx + 384,  0, 3);
        }

        // acc[0..1]=r(s0,s1) acc[2..3]=z(s0,s1) acc[4..5]=n_x acc[6..7]=n_h
        v8f acc[8] = {};

        const float* __restrict__ xrow = x + ((size_t)t * B_DIM + mA) * H_DIM;
        const bf16*  __restrict__ hrow = hrd + (size_t)mA * H_DIM;

        // ---- K-half 1: x contribution (K = 0..511) -----------------------
        #pragma unroll 4
        for (int kt = 0; kt < 16; ++kt) {
            const int c0 = kt * 32 + 8 * lh;          // A chunk offsets (ISA layout)
            v16bf a;
            {
                const f32x4* p0 = (const f32x4*)(xrow + c0);
                const f32x4* p1 = (const f32x4*)(xrow + c0 + 16);
                f32x4 q0 = p0[0], q1 = p0[1], q2 = p1[0], q3 = p1[1];
                #pragma unroll
                for (int i = 0; i < 4; ++i) {
                    a[i]      = (bf16)q0.v[i];
                    a[4 + i]  = (bf16)q1.v[i];
                    a[8 + i]  = (bf16)q2.v[i];
                    a[12 + i] = (bf16)q3.v[i];
                }
            }
            const int koff = kt * 32 + 8 * lh;
            #pragma unroll
            for (int tn = 0; tn < 6; ++tn) {
                const bf16* bp = &ldsB[(size_t)(tn * 16 + l16) * LDS_ROWSTRIDE + koff];
                bfx8 b0 = *(const bfx8*)bp;
                bfx8 b1 = *(const bfx8*)(bp + 16);
                v16bf b;
                #pragma unroll
                for (int i = 0; i < 8; ++i) { b[i] = b0.v[i]; b[8 + i] = b1.v[i]; }
                acc[tn] = __builtin_amdgcn_wmma_f32_16x16x32_bf16(
                    false, a, false, b, (short)0, acc[tn], false, false);
            }
        }

        // ---- K-half 2: h contribution (K = 512..1023), reset-masked ------
        #pragma unroll 4
        for (int kt = 0; kt < 16; ++kt) {
            const int hk = kt * 32 + 8 * lh;
            bfx8 q0 = *(const bfx8*)(hrow + hk);
            bfx8 q1 = *(const bfx8*)(hrow + hk + 16);
            v16bf a;
            #pragma unroll
            for (int i = 0; i < 8; ++i) {
                a[i]     = rA ? zb : q0.v[i];         // v_cndmask, no divergence
                a[8 + i] = rA ? zb : q1.v[i];
            }
            const int koff = (16 + kt) * 32 + 8 * lh;
            #pragma unroll
            for (int tn = 0; tn < 6; ++tn) {
                const bf16* bp = &ldsB[(size_t)(tn * 16 + l16) * LDS_ROWSTRIDE + koff];
                bfx8 b0 = *(const bfx8*)bp;
                bfx8 b1 = *(const bfx8*)(bp + 16);
                v16bf b;
                #pragma unroll
                for (int i = 0; i < 8; ++i) { b[i] = b0.v[i]; b[8 + i] = b1.v[i]; }
                const int ai = (tn < 4) ? tn : (2 + tn);   // n-gate -> acc[6],acc[7]
                acc[ai] = __builtin_amdgcn_wmma_f32_16x16x32_bf16(
                    false, a, false, b, (short)0, acc[ai], false, false);
            }
        }

        // ---- Gates + carry update (D layout: M = v + 8*lh, N = l16) ------
        #pragma unroll
        for (int s = 0; s < 2; ++s) {
            const int j = jbase + 16 * s + l16;
            #pragma unroll
            for (int v = 0; v < 8; ++v) {
                const float r = sigf(acc[0 + s][v] + bi_r[s]);
                const float z = sigf(acc[2 + s][v] + bi_z[s]);
                const float n = tanhf(acc[4 + s][v] + bi_n[s] +
                                      r * (acc[6 + s][v] + bhn[s]));
                const bool mres = ((mbits >> (8 * v)) & 0xffULL) != 0ULL;
                const float hp = mres ? 0.0f : hc[s][v];
                const float hn = (1.0f - z) * n + z * hp;
                hc[s][v] = hn;
                const int m = 16 * w + 8 * lh + v;
                out[YS_OFF + ((size_t)t * B_DIM + m) * H_DIM + j] = hn;   // ys
                hwr[(size_t)m * H_DIM + j] = (bf16)hn;                    // carry
            }
        }

        // ---- Inter-WG barrier (release h stores / acquire for next read) -
        if (t + 1 < T_STEPS) {
            __threadfence();
            __syncthreads();
            if (tid == 0) {
                __hip_atomic_fetch_add(ctr, 1u, __ATOMIC_RELEASE,
                                       __HIP_MEMORY_SCOPE_AGENT);
                const unsigned target = (unsigned)(NWG * (t + 1));
                while (__hip_atomic_load(ctr, __ATOMIC_ACQUIRE,
                                         __HIP_MEMORY_SCOPE_AGENT) < target)
                    __builtin_amdgcn_s_sleep(1);
            }
            __syncthreads();
            __threadfence();
        }
    }

    // ---- h_final ---------------------------------------------------------
    #pragma unroll
    for (int s = 0; s < 2; ++s) {
        const int j = jbase + 16 * s + l16;
        #pragma unroll
        for (int v = 0; v < 8; ++v) {
            const int m = 16 * w + 8 * lh + v;
            out[(size_t)m * H_DIM + j] = hc[s][v];
        }
    }
}

// ---------------------------------------------------------------------------
extern "C" void kernel_launch(void* const* d_in, const int* in_sizes, int n_in,
                              void* d_out, int out_size, void* d_ws, size_t ws_size,
                              hipStream_t stream) {
    const float*         x      = (const float*)d_in[0];
    const unsigned char* resets = (const unsigned char*)d_in[1];  // jnp bool -> 1 byte
    const float*         Wi     = (const float*)d_in[2];
    const float*         bi     = (const float*)d_in[3];
    const float*         Wh     = (const float*)d_in[4];
    const float*         bh_n   = (const float*)d_in[5];

    char* ws = (char*)d_ws;
    unsigned* ctr = (unsigned*)ws;                                   // 256 B reserved
    bf16* h0 = (bf16*)(ws + 256);                                    // 128 KB
    bf16* h1 = (bf16*)(ws + 256 + (size_t)B_DIM * H_DIM * 2);        // 128 KB
    bf16* Wt = (bf16*)(ws + 256 + 2 * (size_t)B_DIM * H_DIM * 2);    // 3 MB packed W^T

    gru_pack_kernel<<<512, 256, 0, stream>>>(Wi, Wh, Wt, h0, h1, ctr);
    gru_scan_kernel<<<NWG, WG_THREADS, LDS_BYTES, stream>>>(
        x, resets, bi, bh_n, Wt, h0, h1, ctr, (float*)d_out);
}
